// CGCNN_78683800863379
// MI455X (gfx1250) — matrix operand order repaired and checked by reading, not compiled, not run on hardware
//
#include <hip/hip_runtime.h>
#include <hip/hip_fp16.h>

// ---------------- problem constants ----------------
#define NN 20000
#define EE 320000
#define GG 128
#define DD 64     // hidden dim
#define DE 32     // rbf dim
#define FA 92
#define FF 128
#define INZ 160   // 2*DD + DE
#define TE 256    // edge rows per GEMM block (8 waves x 32 rows)

typedef __attribute__((ext_vector_type(16))) _Float16 v16h;
typedef __attribute__((ext_vector_type(8)))  _Float16 v8h;
typedef __attribute__((ext_vector_type(8)))  float    v8f;

__device__ __forceinline__ float sp_f(float x) {
    // numerically-stable softplus, matches jax.nn.softplus
    return x > 0.f ? x + log1pf(expf(-x)) : log1pf(expf(x));
}
__device__ __forceinline__ float sigm(float x) { return 1.f / (1.f + expf(-x)); }

// ---------------------------------------------------------------------------
// Pack Wi/Wu (fp32 [3][160][64]) into f16 WMMA B-fragment layout:
// [l][w][kt][nt][lane][j]; col=lane&15, h=lane>>4,
//   K = kt*32 + (j<8 ? 8h + j : 16 + 8h + (j-8))
// (ISA 7.12.2: B rows striped across lanes; lanes 0-15 hold K=0..7,16..23,
//  lanes 16-31 hold K=8..15,24..31 for a 32x16 f16 B tile)
// ---------------------------------------------------------------------------
__global__ void pack_weights_kernel(const float* __restrict__ Wi,
                                    const float* __restrict__ Wu,
                                    _Float16* __restrict__ packed) {
    int idx = blockIdx.x * blockDim.x + threadIdx.x;
    const int total = 3 * 2 * 5 * 4 * 32 * 16;
    if (idx >= total) return;
    int j = idx & 15;       int t = idx >> 4;
    int lane = t & 31;      t >>= 5;
    int nt = t & 3;         t >>= 2;
    int kt = t % 5;         t /= 5;
    int w = t & 1;          int l = t >> 1;
    int col = nt * 16 + (lane & 15);
    int k = kt * 32 + ((lane < 16) ? 0 : 8) + ((j < 8) ? j : 8 + j);
    const float* W = (w == 0) ? Wi : Wu;
    packed[idx] = (_Float16)W[(l * INZ + k) * DD + col];
}

// ---------------------------------------------------------------------------
// RBF expansion of bond lengths -> f16 [E,32]
// centers = linspace(0,8,32); gamma = (De-1)/8
// ---------------------------------------------------------------------------
__global__ void rbf_kernel(const float* __restrict__ bl, _Float16* __restrict__ rbf) {
    int idx = blockIdx.x * blockDim.x + threadIdx.x;
    if (idx >= EE * DE) return;
    int c = idx & (DE - 1);
    int e = idx >> 5;
    float center = c * (8.f / (DE - 1));
    float gamma  = (DE - 1) / 8.f;
    float d = bl[e] - center;
    rbf[idx] = (_Float16)expf(-gamma * d * d);
}

// ---------------------------------------------------------------------------
// h0 = atom_features @ emb_W + emb_b   ([20000,92]x[92,64]) ; also f16 copy
// ---------------------------------------------------------------------------
__global__ void embed_kernel(const float* __restrict__ af,
                             const float* __restrict__ W,
                             const float* __restrict__ b,
                             float* __restrict__ h, _Float16* __restrict__ h16) {
    int idx = blockIdx.x * blockDim.x + threadIdx.x;
    if (idx >= NN * DD) return;
    int d = idx & (DD - 1);
    int n = idx >> 6;
    const float* row = af + n * FA;
    float acc = b[d];
#pragma unroll 4
    for (int f = 0; f < FA; ++f) acc += row[f] * W[f * DD + d];
    h[idx] = acc;
    h16[idx] = (_Float16)acc;
}

// ---------------------------------------------------------------------------
// Edge GEMM: z[E,160](f16, gathered to LDS) x {Wi,Wu}[160,64](f16, LDS) via
// v_wmma_f32_16x16x32_f16. Each wave owns a 32-edge x 64-col tile for BOTH
// weight matrices: each B fragment feeds 2 WMMAs (two M tiles), i.e.
// 5 K-tiles x 4 N-tiles x 2 matrices x 2 M-tiles = 80 WMMAs per wave.
// Biases bi/bu omitted: constant-per-column shifts cancel under BatchNorm.
// ---------------------------------------------------------------------------
__global__ __launch_bounds__(256) void edge_gemm_kernel(
    const _Float16* __restrict__ h16,    // [N,64]
    const _Float16* __restrict__ rbf16,  // [E,32]
    const int* __restrict__ src, const int* __restrict__ dst,
    const _Float16* __restrict__ packedW,  // this layer: [2][5][4][32][16]
    float* __restrict__ yi, float* __restrict__ yu) {
    extern __shared__ char smem[];
    _Float16* zs = (_Float16*)smem;             // 256 rows x 160 halves = 81920 B
    _Float16* ws = (_Float16*)(smem + 81920);   // 2*5*4*512 halves      = 40960 B

    const int t  = threadIdx.x;
    const int e0 = blockIdx.x * TE;

    // --- stage weights into LDS (fragment order, straight b128 copy) ---
    {
        const uint4* g = (const uint4*)packedW;
        uint4* l = (uint4*)ws;
#pragma unroll
        for (int i = 0; i < 10; ++i) l[t + i * 256] = g[t + i * 256];
    }
    // --- gather z tile: cols 0-63 h[src], 64-127 h[dst], 128-159 rbf ---
#pragma unroll
    for (int phase = 0; phase < 2; ++phase) {
        const int* idxp = phase ? dst : src;
        const int colbase = phase ? DD : 0;
#pragma unroll
        for (int i = 0; i < 8; ++i) {
            int idx = i * 256 + t;          // 256 rows x 8 chunks of 8 halves
            int row = idx >> 3, ch = idx & 7;
            int e = e0 + row; if (e >= EE) e = EE - 1;
            int node = idxp[e];
            const uint4* g = (const uint4*)(h16 + (size_t)node * DD);
            *(uint4*)(zs + row * INZ + colbase + ch * 8) = g[ch];
        }
    }
#pragma unroll
    for (int i = 0; i < 4; ++i) {
        int idx = i * 256 + t;              // 256 rows x 4 chunks of 8 halves
        int row = idx >> 2, ch = idx & 3;
        int e = e0 + row; if (e >= EE) e = EE - 1;
        const uint4* g = (const uint4*)(rbf16 + (size_t)e * DE);
        *(uint4*)(zs + row * INZ + 2 * DD + ch * 8) = g[ch];
    }
    __syncthreads();

    const int wv    = t >> 5;
    const int lane  = t & 31;
    const int mrow  = lane & 15;
    const int kbase = (lane < 16) ? 0 : 8;  // A-frag K split per ISA 7.12.2
    const _Float16* zrow0 = zs + (wv * 32 + mrow) * INZ;        // M tile 0
    const _Float16* zrow1 = zrow0 + 16 * INZ;                    // M tile 1

    v8f acc_i[2][4] = {}, acc_u[2][4] = {};

#pragma unroll
    for (int kt = 0; kt < 5; ++kt) {
        union { v16h v; v8h h[2]; } a0, a1;
        a0.h[0] = *(const v8h*)(zrow0 + kt * 32 + kbase);        // K = base..base+7
        a0.h[1] = *(const v8h*)(zrow0 + kt * 32 + 16 + kbase);   // K = 16+base..
        a1.h[0] = *(const v8h*)(zrow1 + kt * 32 + kbase);
        a1.h[1] = *(const v8h*)(zrow1 + kt * 32 + 16 + kbase);
#pragma unroll
        for (int nt = 0; nt < 4; ++nt) {
            union { v16h v; v8h h[2]; } bi, bu;
            const _Float16* wpi = ws + ((0 * 5 + kt) * 4 + nt) * 512 + lane * 16;
            const _Float16* wpu = ws + ((1 * 5 + kt) * 4 + nt) * 512 + lane * 16;
            bi.h[0] = *(const v8h*)wpi;  bi.h[1] = *(const v8h*)(wpi + 8);
            bu.h[0] = *(const v8h*)wpu;  bu.h[1] = *(const v8h*)(wpu + 8);
            acc_i[0][nt] = __builtin_amdgcn_wmma_f32_16x16x32_f16(
                false, a0.v, false, bi.v, (short)0, acc_i[0][nt], false, false);
            acc_i[1][nt] = __builtin_amdgcn_wmma_f32_16x16x32_f16(
                false, a1.v, false, bi.v, (short)0, acc_i[1][nt], false, false);
            acc_u[0][nt] = __builtin_amdgcn_wmma_f32_16x16x32_f16(
                false, a0.v, false, bu.v, (short)0, acc_u[0][nt], false, false);
            acc_u[1][nt] = __builtin_amdgcn_wmma_f32_16x16x32_f16(
                false, a1.v, false, bu.v, (short)0, acc_u[1][nt], false, false);
        }
    }

    // --- store: C/D layout lane%16 = col, VGPR v -> row v (+8 for lanes>=16)
    const int col  = lane & 15;
    const int radd = (lane < 16) ? 0 : 8;
#pragma unroll
    for (int mt = 0; mt < 2; ++mt) {
#pragma unroll
        for (int nt = 0; nt < 4; ++nt) {
#pragma unroll
            for (int v = 0; v < 8; ++v) {
                int e = e0 + wv * 32 + mt * 16 + radd + v;
                if (e < EE) {
                    yi[(size_t)e * DD + nt * 16 + col] = acc_i[mt][nt][v];
                    yu[(size_t)e * DD + nt * 16 + col] = acc_u[mt][nt][v];
                }
            }
        }
    }
}

// ---------------------------------------------------------------------------
// Per-column sum / sum-of-squares over a [rows,64] matrix (BN batch stats).
// ---------------------------------------------------------------------------
__global__ void colstats_kernel(const float* __restrict__ x, int rows,
                                float* __restrict__ sum, float* __restrict__ sumsq) {
    int c = threadIdx.x & 63;
    int rgrp = blockIdx.x * (blockDim.x >> 6) + (threadIdx.x >> 6);
    int rstride = gridDim.x * (blockDim.x >> 6);
    float s = 0.f, q = 0.f;
    for (int r = rgrp; r < rows; r += rstride) {
        float v = x[(size_t)r * DD + c];
        s += v; q += v * v;
    }
    atomicAdd(&sum[c], s);
    atomicAdd(&sumsq[c], q);
}

// ---------------------------------------------------------------------------
// gate = sigmoid(BN(yi)); upd = softplus(BN(yu)); m[dst] += gate*upd
// st: [0:64) sum_i [64:128) sq_i [128:192) sum_u [192:256) sq_u
// ---------------------------------------------------------------------------
__global__ void edge_apply_kernel(const float* __restrict__ yi,
                                  const float* __restrict__ yu,
                                  const int* __restrict__ dst,
                                  const float* __restrict__ gi, const float* __restrict__ bbi,
                                  const float* __restrict__ gu, const float* __restrict__ bbu,
                                  const float* __restrict__ st,
                                  float* __restrict__ m) {
    int idx = blockIdx.x * blockDim.x + threadIdx.x;
    if (idx >= EE * DD) return;
    int c = idx & 63;
    int e = idx >> 6;
    const float invE = 1.f / EE;
    float mu_i = st[c] * invE,        var_i = st[64 + c]  * invE - mu_i * mu_i;
    float mu_u = st[128 + c] * invE,  var_u = st[192 + c] * invE - mu_u * mu_u;
    float xi = gi[c] * (yi[idx] - mu_i) * rsqrtf(var_i + 1e-5f) + bbi[c];
    float xu = gu[c] * (yu[idx] - mu_u) * rsqrtf(var_u + 1e-5f) + bbu[c];
    atomicAdd(&m[(size_t)dst[e] * DD + c], sigm(xi) * sp_f(xu));
}

// ---------------------------------------------------------------------------
// h = softplus(h + BN(m)); refresh f16 copy; last layer: pool per graph.
// stm: [0:64) sum  [64:128) sumsq  over N rows
// ---------------------------------------------------------------------------
__global__ void node_update_kernel(float* __restrict__ h, _Float16* __restrict__ h16,
                                   const float* __restrict__ m,
                                   const float* __restrict__ gn, const float* __restrict__ bbn,
                                   const float* __restrict__ stm,
                                   const int* __restrict__ graph_id,
                                   float* __restrict__ pool_sum, float* __restrict__ pool_cnt,
                                   int last) {
    int idx = blockIdx.x * blockDim.x + threadIdx.x;
    if (idx >= NN * DD) return;
    int c = idx & 63;
    int n = idx >> 6;
    const float invN = 1.f / NN;
    float mu = stm[c] * invN, var = stm[64 + c] * invN - mu * mu;
    float bn = gn[c] * (m[idx] - mu) * rsqrtf(var + 1e-5f) + bbn[c];
    float hn = sp_f(h[idx] + bn);
    h[idx] = hn;
    h16[idx] = (_Float16)hn;
    if (last) {
        int g = graph_id[n];
        atomicAdd(&pool_sum[g * DD + c], hn);
        if (c == 0) atomicAdd(&pool_cnt[g], 1.f);
    }
}

// ---------------------------------------------------------------------------
// Head: feats=sp(mean); feats=sp(feats@fcW+fcb); feats=sp(feats); out=feats@outW+outb
// one thread per graph (128 graphs, ~8.5K MACs each -> trivial)
// ---------------------------------------------------------------------------
__global__ void head_kernel(const float* __restrict__ pool_sum,
                            const float* __restrict__ pool_cnt,
                            const float* __restrict__ fcW, const float* __restrict__ fcb,
                            const float* __restrict__ outW, const float* __restrict__ outb,
                            float* __restrict__ out) {
    int g = blockIdx.x * blockDim.x + threadIdx.x;
    if (g >= GG) return;
    float x[DD];
    float cnt = fmaxf(pool_cnt[g], 1.f);
#pragma unroll
    for (int d = 0; d < DD; ++d) x[d] = sp_f(pool_sum[g * DD + d] / cnt);
    float acc = outb[0];
    for (int j = 0; j < FF; ++j) {
        float y = fcb[j];
#pragma unroll 8
        for (int d = 0; d < DD; ++d) y += x[d] * fcW[d * FF + j];
        acc += sp_f(sp_f(y)) * outW[j];
    }
    out[g] = acc;
}

// ---------------------------------------------------------------------------
extern "C" void kernel_launch(void* const* d_in, const int* in_sizes, int n_in,
                              void* d_out, int out_size, void* d_ws, size_t ws_size,
                              hipStream_t stream) {
    (void)in_sizes; (void)n_in; (void)out_size; (void)ws_size;
    const float* atom = (const float*)d_in[0];
    const float* bond = (const float*)d_in[1];
    const int*   src  = (const int*)d_in[2];
    const int*   dst  = (const int*)d_in[3];
    const int*   gid  = (const int*)d_in[4];
    const float* embW = (const float*)d_in[5];
    const float* embB = (const float*)d_in[6];
    const float* Wi   = (const float*)d_in[7];
    /* d_in[8] = bi : cancels under BN */
    const float* gi   = (const float*)d_in[9];
    const float* bbi  = (const float*)d_in[10];
    const float* Wu   = (const float*)d_in[11];
    /* d_in[12] = bu : cancels under BN */
    const float* gu   = (const float*)d_in[13];
    const float* bbu  = (const float*)d_in[14];
    const float* gn   = (const float*)d_in[15];
    const float* bbn  = (const float*)d_in[16];
    const float* fcW  = (const float*)d_in[17];
    const float* fcb  = (const float*)d_in[18];
    const float* outW = (const float*)d_in[19];
    const float* outb = (const float*)d_in[20];
    float* out = (float*)d_out;

    // ---- workspace carve-up (256B aligned slices) ----
    char* p = (char*)d_ws;
    auto carve = [&](size_t bytes) -> char* {
        char* r = p; p += (bytes + 255) & ~(size_t)255; return r;
    };
    float*    h      = (float*)   carve((size_t)NN * DD * 4);
    _Float16* h16    = (_Float16*)carve((size_t)NN * DD * 2);
    _Float16* rbf    = (_Float16*)carve((size_t)EE * DE * 2);
    _Float16* packed = (_Float16*)carve((size_t)3 * 2 * 5 * 4 * 32 * 16 * 2);
    float*    yi     = (float*)   carve((size_t)EE * DD * 4);
    float*    yu     = (float*)   carve((size_t)EE * DD * 4);
    float*    m      = (float*)   carve((size_t)NN * DD * 4);
    float*    st     = (float*)   carve(384 * 4);  // sum_i,sq_i,sum_u,sq_u,sum_m,sq_m
    float*    pool   = (float*)   carve((size_t)GG * DD * 4);
    float*    cnt    = (float*)   carve((size_t)GG * 4);

    // ---- preprocessing ----
    pack_weights_kernel<<<(3 * 2 * 5 * 4 * 32 * 16 + 255) / 256, 256, 0, stream>>>(Wi, Wu, packed);
    rbf_kernel<<<((size_t)EE * DE + 255) / 256, 256, 0, stream>>>(bond, rbf);
    embed_kernel<<<((size_t)NN * DD + 255) / 256, 256, 0, stream>>>(atom, embW, embB, h, h16);
    hipMemsetAsync(pool, 0, (size_t)GG * DD * 4, stream);
    hipMemsetAsync(cnt, 0, (size_t)GG * 4, stream);

    const size_t lds_bytes = 81920 + 40960;  // z tile + both weight matrices
    for (int l = 0; l < 3; ++l) {
        hipMemsetAsync(m, 0, (size_t)NN * DD * 4, stream);
        hipMemsetAsync(st, 0, 384 * 4, stream);
        edge_gemm_kernel<<<EE / TE, 256, lds_bytes, stream>>>(
            h16, rbf, src, dst, packed + (size_t)l * 2 * 5 * 4 * 32 * 16, yi, yu);
        colstats_kernel<<<256, 256, 0, stream>>>(yi, EE, st, st + 64);
        colstats_kernel<<<256, 256, 0, stream>>>(yu, EE, st + 128, st + 192);
        edge_apply_kernel<<<((size_t)EE * DD + 255) / 256, 256, 0, stream>>>(
            yi, yu, dst, gi + l * DD, bbi + l * DD, gu + l * DD, bbu + l * DD, st, m);
        colstats_kernel<<<256, 256, 0, stream>>>(m, NN, st + 256, st + 320);
        node_update_kernel<<<((size_t)NN * DD + 255) / 256, 256, 0, stream>>>(
            h, h16, m, gn + l * DD, bbn + l * DD, st + 256, gid, pool, cnt, l == 2);
    }
    head_kernel<<<1, 128, 0, stream>>>(pool, cnt, fcW, fcb, outW, outb, out);
}
